// CRF_2671469658740
// MI455X (gfx1250) — matrix operand order repaired and checked by reading, not compiled, not run on hardware
//
#include <hip/hip_runtime.h>
#include <hip/hip_bf16.h>
#include <stdint.h>

typedef __attribute__((ext_vector_type(16))) __bf16 v16bf;
typedef __attribute__((ext_vector_type(8)))  float  v8f;

#define N_     2048
#define D_     512
#define TM     16          // rows per workgroup strip
#define PSTR   2068        // padded f32 stride of P/Wsym strip (8*PSTR % 64 == 32)
#define ASTR   520         // padded bf16 stride of staged A rows (1040B, 16B aligned)

// LDS partition (bytes)
#define OFF_STRIP 0
#define SZ_STRIP  (TM * PSTR * 4)          // 132352
#define OFF_A     (OFF_STRIP + SZ_STRIP)
#define SZ_A      (TM * ASTR * 2)          // 16640
#define OFF_LG    (OFF_A + SZ_A)
#define SZ_LG     (N_ * 4)                 // 8192
#define SMEM_SZ   (OFF_LG + SZ_LG)         // 157184 B (< 160KB -> 2 WGs/WGP)

// --- Kernel 1: row-normalize feats (f32) -> Ahat (bf16) in workspace ---
__global__ void crf_normalize(const float* __restrict__ feats,
                              __bf16* __restrict__ ahat) {
  const int row = blockIdx.x;                 // B*N rows
  const float* f = feats + (size_t)row * D_;
  const int t = threadIdx.x;                  // 256 threads
  float v0 = f[t], v1 = f[t + 256];
  float ss = v0 * v0 + v1 * v1;
  #pragma unroll
  for (int m = 16; m >= 1; m >>= 1) ss += __shfl_xor(ss, m, 32);
  __shared__ float red[8];
  if ((t & 31) == 0) red[t >> 5] = ss;
  __syncthreads();
  float tot = red[0] + red[1] + red[2] + red[3] +
              red[4] + red[5] + red[6] + red[7];
  float rn = rsqrtf(tot);
  ahat[(size_t)row * D_ + t]       = (__bf16)(v0 * rn);
  ahat[(size_t)row * D_ + t + 256] = (__bf16)(v1 * rn);
}

__device__ __forceinline__ float sigmoidf_(float x) {
  return 1.0f / (1.0f + __expf(-x));
}

// --- Kernel 2: fused  P = (Ahat Ahat^T) .* Wsym  (WMMA) -> 10 CRF iters -> out ---
__global__ void __launch_bounds__(256)
crf_fused(const __bf16* __restrict__ ahat, const float* __restrict__ logits,
          const float* __restrict__ W, float* __restrict__ out) {
  extern __shared__ char smem[];
  float*  strip = (float*)(smem + OFF_STRIP);   // [TM][PSTR]  Wsym -> P
  __bf16* lds_a = (__bf16*)(smem + OFF_A);      // [TM][ASTR]  A strip (bf16)
  float*  lg    = (float*)(smem + OFF_LG);      // [N] logits

  const int i0 = blockIdx.x * TM;               // row strip base
  const int b  = blockIdx.y;                    // batch
  const int t  = threadIdx.x;                   // 256 threads = 8 waves
  const int w    = t >> 5;
  const int lane = t & 31;

  // ---- preload pass A: 0.5*W rows -> strip ; stage A rows ; logits ----
  for (int e = t; e < TM * N_; e += 256) {      // row-major: coalesced
    int iL = e >> 11, j = e & (N_ - 1);
    strip[iL * PSTR + j] = 0.5f * W[(size_t)(i0 + iL) * N_ + j];
  }
  {
    const uint32_t* asrc =
        (const uint32_t*)(ahat + (size_t)(b * N_ + i0) * D_);
    uint32_t* adst = (uint32_t*)lds_a;
    for (int e = t; e < TM * (D_ / 2); e += 256) {
      int iL = e >> 8, c = e & 255;
      adst[iL * (ASTR / 2) + c] = asrc[iL * (D_ / 2) + c];
    }
  }
  for (int j = t; j < N_; j += 256) lg[j] = logits[b * N_ + j];
  __syncthreads();

  // ---- preload pass B: += 0.5*W columns (transposed, 64B-coalesced groups) ----
  for (int e = t; e < TM * N_; e += 256) {
    int j = e >> 4, iL = e & (TM - 1);
    strip[iL * PSTR + j] += 0.5f * W[(size_t)j * N_ + (i0 + iL)];
  }
  __syncthreads();

  // ---- GEMM: each wave does 16 j-tiles; K-loop of 16 bf16 WMMAs ----
  const __bf16* abase = ahat + (size_t)b * N_ * D_;
  const __bf16* arow  = lds_a + (lane & 15) * ASTR + ((lane & 16) ? 8 : 0);
  const int ib = (lane & 16) ? 8 : 0;

  for (int jt = w; jt < N_ / 16; jt += 8) {
    const int j0 = jt * 16;
    const __bf16* rowj =
        abase + (size_t)(j0 + (lane & 15)) * D_ + ((lane & 16) ? 8 : 0);
    v8f c = {0.f, 0.f, 0.f, 0.f, 0.f, 0.f, 0.f, 0.f};
    #pragma unroll 4
    for (int k0 = 0; k0 < D_; k0 += 32) {
      union { v16bf v; uint4 q[2]; } A, B;
      A.q[0] = *(const uint4*)(arow + k0);        // K 0-7  (or 8-15)
      A.q[1] = *(const uint4*)(arow + k0 + 16);   // K 16-23 (or 24-31)
      B.q[0] = *(const uint4*)(rowj + k0);        // K 0-7  (or 8-15)
      B.q[1] = *(const uint4*)(rowj + k0 + 16);
      c = __builtin_amdgcn_wmma_f32_16x16x32_bf16(
              false, A.v, false, B.v, (short)0, c, false, false);
    }
    // epilogue: P = cos * Wsym, in place in LDS
    const int jcol = j0 + (lane & 15);
    #pragma unroll
    for (int r = 0; r < 8; ++r) {
      float* p = &strip[(ib + r) * PSTR + jcol];
      *p = c[r] * (*p);
    }
  }
  __syncthreads();

  // ---- 10 CRF iterations, P resident in LDS; 16 lanes per row ----
  const int iL = t >> 4, sub = t & 15;
  const float* prow = strip + iL * PSTR;
  float acc = 0.f;
  for (int j = sub; j < N_; j += 16) acc += prow[j];
  #pragma unroll
  for (int m = 8; m >= 1; m >>= 1) acc += __shfl_xor(acc, m, 32);
  float s = 0.5f * acc;                         // iteration 1 (Q0 = 0.5)
  for (int it = 0; it < 9; ++it) {              // iterations 2..10
    acc = 0.f;
    for (int j = sub; j < N_; j += 16)
      acc += prow[j] * sigmoidf_(lg[j] + s);
    #pragma unroll
    for (int m = 8; m >= 1; m >>= 1) acc += __shfl_xor(acc, m, 32);
    s = acc;
  }

  // ---- output: Q[b, i0+iL, j] = sigmoid(logits[j] + s) ----
  float* orow = out + ((size_t)b * N_ + (i0 + iL)) * (size_t)N_;
  for (int j = sub; j < N_; j += 16)
    orow[j] = sigmoidf_(lg[j] + s);
}

extern "C" void kernel_launch(void* const* d_in, const int* in_sizes, int n_in,
                              void* d_out, int out_size, void* d_ws, size_t ws_size,
                              hipStream_t stream) {
  const float* feats  = (const float*)d_in[0];  // [B,N,D] f32
  const float* logits = (const float*)d_in[1];  // [B,N,1] f32
  const float* W      = (const float*)d_in[2];  // [1,N,N] f32
  float* out = (float*)d_out;                   // [B,N,N,1] f32
  const int B = in_sizes[1] / N_;               // 4

  __bf16* ahat = (__bf16*)d_ws;                 // [B,N,D] bf16 (8 MiB)

  crf_normalize<<<dim3(B * N_), 256, 0, stream>>>(feats, ahat);
  crf_fused<<<dim3(N_ / TM, B), 256, SMEM_SZ, stream>>>(ahat, logits, W, out);
}
